// Rs_GCNU_29137058136126
// MI455X (gfx1250) — matrix operand order repaired and checked by reading, not compiled, not run on hardware
//
#include <hip/hip_runtime.h>
#include <hip/hip_bf16.h>

#define B_  4
#define N_  4096
#define C_  1024
#define IC_ 512
#define EPS_ 1e-5f

typedef __bf16 bf16;
typedef __attribute__((ext_vector_type(16))) __bf16 v16bf;
typedef __attribute__((ext_vector_type(8)))  float  v8f;
typedef __attribute__((ext_vector_type(4)))  unsigned int v4u;
typedef __attribute__((ext_vector_type(8)))  int v8i;
typedef __attribute__((ext_vector_type(4)))  int v4i;

#if __has_builtin(__builtin_amdgcn_tensor_load_to_lds)
#define HAVE_TDM 1
#endif

// ---------------------------------------------------------------------------
// WMMA: D = A(16x32 bf16) * B(32x16 bf16) + C(16x16 f32)
// ---------------------------------------------------------------------------
__device__ __forceinline__ v8f wmma_bf16(v16bf a, v16bf b, v8f c) {
  return __builtin_amdgcn_wmma_f32_16x16x32_bf16(
      false, a, false, b, (short)0, c, false, false);
}

// A fragment (16x32 bf16): lanes 0-15 row m K 0..15; lanes 16-31 row m
// K 16..31.  Row-major source => contiguous per-lane runs.
__device__ __forceinline__ v16bf frag_a_bf16(const bf16* A, int lda, int m, int hi, int kbase) {
  const bf16* p = A + (size_t)m * lda + kbase + hi * 16;
  v16bf r;
#pragma unroll
  for (int j = 0; j < 16; ++j) r[j] = p[j];
  return r;
}
// B fragment (32x16): lane = column n; lanes 0-15 K 0..15, lanes 16-31 K 16..31.
// Source row n of a row-major matrix (B = W^T) => 16 contiguous values.
__device__ __forceinline__ v16bf frag_b_bf16(const bf16* W, int ldw, int n, int hi, int kbase) {
  const bf16* p = W + (size_t)n * ldw + kbase + hi * 16;
  v16bf r;
#pragma unroll
  for (int j = 0; j < 16; ++j) r[j] = p[j];
  return r;
}

// ---------------------------------------------------------------------------
// Tensor Data Mover: 2D tile (tile_k elems x tile_rows) of 2-byte data,
// row stride = stride0 elems, into LDS at lds_off.  D# per CDNA5 ISA §8.
// ---------------------------------------------------------------------------
__device__ __forceinline__ unsigned lds_off_of(const void* p) {
  return (unsigned)(unsigned long long)p;   // LDS aperture: low 32 bits = byte offset
}

__device__ __forceinline__ void tdm_load_2d(unsigned lds_off, unsigned long long gaddr,
                                            unsigned tile_k, unsigned tile_rows,
                                            unsigned tensor_d0, unsigned tensor_d1,
                                            unsigned long long stride0) {
#ifdef HAVE_TDM
  v4u g0;
  g0[0] = 1u;                                                    // count=1 (valid user D#)
  g0[1] = lds_off;                                               // lds_addr
  g0[2] = (unsigned)(gaddr & 0xFFFFFFFFull);                     // global_addr[31:0]
  g0[3] = (unsigned)((gaddr >> 32) & 0x1FFFFFFull) | (2u << 30); // addr[56:32] | type=2
  v8i g1;
  g1[0] = (int)(1u << 16);                                       // data_size=1 (2B), mask=0
  g1[1] = (int)((tensor_d0 & 0xFFFFu) << 16);                    // tensor_dim0[15:0] @ bit48
  g1[2] = (int)(((tensor_d0 >> 16) & 0xFFFFu) |
                ((tensor_d1 & 0xFFFFu) << 16));                  // dim0 hi | dim1 lo
  g1[3] = (int)(((tensor_d1 >> 16) & 0xFFFFu) |
                ((tile_k & 0xFFFFu) << 16));                     // dim1 hi | tile_dim0
  g1[4] = (int)(tile_rows & 0xFFFFu);                            // tile_dim1, tile_dim2=0
  g1[5] = (int)(unsigned)(stride0 & 0xFFFFFFFFull);              // dim0_stride lo
  g1[6] = (int)(unsigned)((stride0 >> 32) & 0xFFFFull);          // dim0_stride hi
  g1[7] = 0;
  v4i g2 = {0, 0, 0, 0};
  v4i g3 = {0, 0, 0, 0};
#if __clang_major__ >= 23
  v8i g4 = {0, 0, 0, 0, 0, 0, 0, 0};
  __builtin_amdgcn_tensor_load_to_lds(g0, g1, g2, g3, g4, 0);
#else
  __builtin_amdgcn_tensor_load_to_lds(g0, g1, g2, g3, 0);
#endif
#else
  (void)lds_off; (void)gaddr; (void)tile_k; (void)tile_rows;
  (void)tensor_d0; (void)tensor_d1; (void)stride0;
#endif
}

// ---------------------------------------------------------------------------
// K0: one-shot fp32 -> bf16 conversion (v and all weight matrices).
// Pays ~100 MB of streaming once to halve GEMM operand bytes and strip the
// v_cvt_pk_bf16_f32 chains out of every WMMA inner loop.
// ---------------------------------------------------------------------------
__global__ __launch_bounds__(256) void k_cvt(
    const float* __restrict__ src, bf16* __restrict__ dst, int n4)
{
  int i = blockIdx.x * blockDim.x + threadIdx.x;
  const int stride = gridDim.x * blockDim.x;
  for (; i < n4; i += stride) {
    const float4 f = ((const float4*)src)[i];
    dst[4 * i + 0] = (bf16)f.x;
    dst[4 * i + 1] = (bf16)f.y;
    dst[4 * i + 2] = (bf16)f.z;
    dst[4 * i + 3] = (bf16)f.w;
  }
}

// ---------------------------------------------------------------------------
// K1: fused projections, 2x2 register blocking (wave tile 32 rows x 32 cols,
// 3 outputs => 12 WMMA per K-step off 2 A + 6 B bf16 fragment loads).
// theta -> row-major (N x IC); phi,g -> transposed (IC x N).
// ---------------------------------------------------------------------------
__global__ __launch_bounds__(256) void k_proj(
    const bf16* __restrict__ vbf,
    const bf16* __restrict__ gw, const float* __restrict__ gb,
    const bf16* __restrict__ tw, const float* __restrict__ tb,
    const bf16* __restrict__ pw, const float* __restrict__ pb,
    bf16* __restrict__ th, bf16* __restrict__ phT, bf16* __restrict__ gvT)
{
  const int lane = threadIdx.x & 31;
  int wid = (blockIdx.x * blockDim.x + threadIdx.x) >> 5;
  const int ti = wid & 15;  wid >>= 4;   // IC/32 = 16
  const int tn = wid & 127; wid >>= 7;   // N/32 = 128
  const int b  = wid;
  const int m = lane & 15, hi = lane >> 4;
  const int row0 = tn * 32, col0 = ti * 32;
  const int c0 = col0 + m, c1 = c0 + 16;

  const bf16* A0 = vbf + (size_t)b * N_ * C_ + (size_t)row0 * C_;
  const bf16* A1 = A0 + (size_t)16 * C_;

  v8f ag[2][2] = {}, at[2][2] = {}, ap[2][2] = {};
  for (int k = 0; k < C_; k += 32) {
    __builtin_prefetch(A0 + (size_t)m * C_ + k + 256, 0, 1);
    v16bf a0 = frag_a_bf16(A0, C_, m, hi, k);
    v16bf a1 = frag_a_bf16(A1, C_, m, hi, k);
    {
      v16bf b0 = frag_b_bf16(gw, C_, c0, hi, k);
      v16bf b1 = frag_b_bf16(gw, C_, c1, hi, k);
      ag[0][0] = wmma_bf16(a0, b0, ag[0][0]); ag[0][1] = wmma_bf16(a0, b1, ag[0][1]);
      ag[1][0] = wmma_bf16(a1, b0, ag[1][0]); ag[1][1] = wmma_bf16(a1, b1, ag[1][1]);
    }
    {
      v16bf b0 = frag_b_bf16(tw, C_, c0, hi, k);
      v16bf b1 = frag_b_bf16(tw, C_, c1, hi, k);
      at[0][0] = wmma_bf16(a0, b0, at[0][0]); at[0][1] = wmma_bf16(a0, b1, at[0][1]);
      at[1][0] = wmma_bf16(a1, b0, at[1][0]); at[1][1] = wmma_bf16(a1, b1, at[1][1]);
    }
    {
      v16bf b0 = frag_b_bf16(pw, C_, c0, hi, k);
      v16bf b1 = frag_b_bf16(pw, C_, c1, hi, k);
      ap[0][0] = wmma_bf16(a0, b0, ap[0][0]); ap[0][1] = wmma_bf16(a0, b1, ap[0][1]);
      ap[1][0] = wmma_bf16(a1, b0, ap[1][0]); ap[1][1] = wmma_bf16(a1, b1, ap[1][1]);
    }
  }

  bf16* thp = th  + (size_t)b * N_ * IC_;
  bf16* php = phT + (size_t)b * IC_ * N_;
  bf16* gvp = gvT + (size_t)b * IC_ * N_;
#pragma unroll
  for (int ct = 0; ct < 2; ++ct) {
    const int col = col0 + ct * 16 + m;
    const float bg = gb[col], bt = tb[col], bp = pb[col];
#pragma unroll
    for (int rt = 0; rt < 2; ++rt) {
      const int rbase = row0 + rt * 16 + hi * 8;
      bf16* pp = php + (size_t)col * N_ + rbase;
      bf16* gp = gvp + (size_t)col * N_ + rbase;
#pragma unroll
      for (int r = 0; r < 8; ++r) {
        thp[(size_t)(rbase + r) * IC_ + col] = (bf16)(at[rt][ct][r] + bt);
        pp[r] = (bf16)(ap[rt][ct][r] + bp);
        gp[r] = (bf16)(ag[rt][ct][r] + bg);
      }
    }
  }
}

// ---------------------------------------------------------------------------
// K2: S = phi^T @ g_v / N  (512x512 per batch, K = 4096).
// 64x64 block tile, A/B panels (64x32 bf16 = 4KB) staged in LDS.
// TDM double-buffered pipeline when available; cooperative loads otherwise.
// ---------------------------------------------------------------------------
__global__ __launch_bounds__(256) void k_S(
    const bf16* __restrict__ phT, const bf16* __restrict__ gvT,
    bf16* __restrict__ ST)
{
  __shared__ bf16 shA[2][64 * 32];
  __shared__ bf16 shB[2][64 * 32];

  const int tid  = threadIdx.x;
  const int lane = tid & 31;
  const int wave = tid >> 5;
  int bi = blockIdx.x;
  const int jb = bi & 7; bi >>= 3;
  const int ib = bi & 7; bi >>= 3;
  const int b  = bi;
  const int i0 = ib * 64, j0 = jb * 64;

  const bf16* Ab = phT + (size_t)b * IC_ * N_ + (size_t)i0 * N_;
  const bf16* Bb = gvT + (size_t)b * IC_ * N_ + (size_t)j0 * N_;

  const int m = lane & 15, hi = lane >> 4;
  const int rt  = wave >> 1;          // row subtile 0..3
  const int ct0 = (wave & 1) * 2;     // col subtiles ct0, ct0+1

  v8f acc0 = {}, acc1 = {};
  const int NSTAGE = N_ / 32;         // 128

#ifdef HAVE_TDM
  if (wave == 0) {
    tdm_load_2d(lds_off_of(shA[0]), (unsigned long long)(size_t)Ab,
                32, 64, N_, IC_, N_);
    tdm_load_2d(lds_off_of(shB[0]), (unsigned long long)(size_t)Bb,
                32, 64, N_, IC_, N_);
    __builtin_amdgcn_s_wait_tensorcnt(0);
  }
  __syncthreads();
  for (int s = 0; s < NSTAGE; ++s) {
    const int cur = s & 1;
    if (wave == 0 && s + 1 < NSTAGE) {
      const unsigned long long koff = (unsigned long long)(s + 1) * 32ull * sizeof(bf16);
      tdm_load_2d(lds_off_of(shA[cur ^ 1]),
                  (unsigned long long)(size_t)Ab + koff, 32, 64, N_, IC_, N_);
      tdm_load_2d(lds_off_of(shB[cur ^ 1]),
                  (unsigned long long)(size_t)Bb + koff, 32, 64, N_, IC_, N_);
    }
    v16bf fa  = frag_a_bf16(shA[cur], 32, rt * 16 + m, hi, 0);
    v16bf fb0 = frag_b_bf16(shB[cur], 32, ct0 * 16 + m, hi, 0);
    v16bf fb1 = frag_b_bf16(shB[cur], 32, ct0 * 16 + 16 + m, hi, 0);
    acc0 = wmma_bf16(fa, fb0, acc0);
    acc1 = wmma_bf16(fa, fb1, acc1);
    if (wave == 0) __builtin_amdgcn_s_wait_tensorcnt(0);
    __syncthreads();
  }
#else
  const int lrow = tid >> 2;          // 0..63
  const int lk   = (tid & 3) * 8;     // 8 bf16 = 16B per thread per panel
  for (int s = 0; s < NSTAGE; ++s) {
    const int k = s * 32;
    __syncthreads();
    *(int4*)&shA[0][lrow * 32 + lk] = *(const int4*)&Ab[(size_t)lrow * N_ + k + lk];
    *(int4*)&shB[0][lrow * 32 + lk] = *(const int4*)&Bb[(size_t)lrow * N_ + k + lk];
    __syncthreads();
    v16bf fa  = frag_a_bf16(shA[0], 32, rt * 16 + m, hi, 0);
    v16bf fb0 = frag_b_bf16(shB[0], 32, ct0 * 16 + m, hi, 0);
    v16bf fb1 = frag_b_bf16(shB[0], 32, ct0 * 16 + 16 + m, hi, 0);
    acc0 = wmma_bf16(fa, fb0, acc0);
    acc1 = wmma_bf16(fa, fb1, acc1);
  }
#endif

  const float scale = 1.0f / (float)N_;
  const int irow = i0 + rt * 16 + hi * 8;
  {
    const int col = j0 + ct0 * 16 + m;
    bf16* sp = ST + (size_t)b * IC_ * IC_ + (size_t)col * IC_ + irow;
#pragma unroll
    for (int r = 0; r < 8; ++r) sp[r] = (bf16)(acc0[r] * scale);
  }
  {
    const int col = j0 + (ct0 + 1) * 16 + m;
    bf16* sp = ST + (size_t)b * IC_ * IC_ + (size_t)col * IC_ + irow;
#pragma unroll
    for (int r = 0; r < 8; ++r) sp[r] = (bf16)(acc1[r] * scale);
  }
}

// ---------------------------------------------------------------------------
// K3: y = theta @ S  (N x IC, K = 512), 2x2 register blocking.
// ---------------------------------------------------------------------------
__global__ __launch_bounds__(256) void k_y(
    const bf16* __restrict__ th, const bf16* __restrict__ ST,
    bf16* __restrict__ y)
{
  const int lane = threadIdx.x & 31;
  int wid = (blockIdx.x * blockDim.x + threadIdx.x) >> 5;
  const int tj = wid & 15;  wid >>= 4;   // IC/32
  const int tn = wid & 127; wid >>= 7;   // N/32
  const int b  = wid;
  const int m = lane & 15, hi = lane >> 4;
  const int row0 = tn * 32, col0 = tj * 32;
  const int c0 = col0 + m, c1 = c0 + 16;

  const bf16* A0 = th + (size_t)b * N_ * IC_ + (size_t)row0 * IC_;
  const bf16* A1 = A0 + (size_t)16 * IC_;
  const bf16* Bm = ST + (size_t)b * IC_ * IC_;

  v8f acc[2][2] = {};
  for (int k = 0; k < IC_; k += 32) {
    __builtin_prefetch(A0 + (size_t)m * IC_ + k + 256, 0, 1);
    v16bf a0 = frag_a_bf16(A0, IC_, m, hi, k);
    v16bf a1 = frag_a_bf16(A1, IC_, m, hi, k);
    v16bf b0 = frag_b_bf16(Bm, IC_, c0, hi, k);
    v16bf b1 = frag_b_bf16(Bm, IC_, c1, hi, k);
    acc[0][0] = wmma_bf16(a0, b0, acc[0][0]); acc[0][1] = wmma_bf16(a0, b1, acc[0][1]);
    acc[1][0] = wmma_bf16(a1, b0, acc[1][0]); acc[1][1] = wmma_bf16(a1, b1, acc[1][1]);
  }

  bf16* yp = y + (size_t)b * N_ * IC_;
#pragma unroll
  for (int rt = 0; rt < 2; ++rt)
#pragma unroll
    for (int ct = 0; ct < 2; ++ct) {
      const int rbase = row0 + rt * 16 + hi * 8;
      const int col = col0 + ct * 16 + m;
#pragma unroll
      for (int r = 0; r < 8; ++r)
        yp[(size_t)(rbase + r) * IC_ + col] = (bf16)acc[rt][ct][r];
    }
}

// ---------------------------------------------------------------------------
// K4: Wy = y @ W_w^T + W_b  (N x C, K = 512), 2x2 register blocking.
// ---------------------------------------------------------------------------
__global__ __launch_bounds__(256) void k_W(
    const bf16* __restrict__ y, const bf16* __restrict__ Ww,
    const float* __restrict__ Wb, float* __restrict__ Wy)
{
  const int lane = threadIdx.x & 31;
  int wid = (blockIdx.x * blockDim.x + threadIdx.x) >> 5;
  const int tc = wid & 31;  wid >>= 5;   // C/32 = 32
  const int tn = wid & 127; wid >>= 7;   // N/32 = 128
  const int b  = wid;
  const int m = lane & 15, hi = lane >> 4;
  const int row0 = tn * 32, col0 = tc * 32;
  const int c0 = col0 + m, c1 = c0 + 16;

  const bf16* A0 = y + (size_t)b * N_ * IC_ + (size_t)row0 * IC_;
  const bf16* A1 = A0 + (size_t)16 * IC_;

  v8f acc[2][2] = {};
  for (int k = 0; k < IC_; k += 32) {
    __builtin_prefetch(A0 + (size_t)m * IC_ + k + 256, 0, 1);
    v16bf a0 = frag_a_bf16(A0, IC_, m, hi, k);
    v16bf a1 = frag_a_bf16(A1, IC_, m, hi, k);
    v16bf b0 = frag_b_bf16(Ww, IC_, c0, hi, k);
    v16bf b1 = frag_b_bf16(Ww, IC_, c1, hi, k);
    acc[0][0] = wmma_bf16(a0, b0, acc[0][0]); acc[0][1] = wmma_bf16(a0, b1, acc[0][1]);
    acc[1][0] = wmma_bf16(a1, b0, acc[1][0]); acc[1][1] = wmma_bf16(a1, b1, acc[1][1]);
  }

  float* wp = Wy + (size_t)b * N_ * C_;
#pragma unroll
  for (int ct = 0; ct < 2; ++ct) {
    const int col = col0 + ct * 16 + m;
    const float bias = Wb[col];
#pragma unroll
    for (int rt = 0; rt < 2; ++rt) {
      const int rbase = row0 + rt * 16 + hi * 8;
#pragma unroll
      for (int r = 0; r < 8; ++r)
        wp[(size_t)(rbase + r) * C_ + col] = acc[rt][ct][r] + bias;
    }
  }
}

// ---------------------------------------------------------------------------
// K5: deterministic per-channel sum/sumsq (one block per channel, LDS tree).
// ---------------------------------------------------------------------------
__global__ __launch_bounds__(256) void k_stats(
    const float* __restrict__ Wy, float* __restrict__ sums,
    float* __restrict__ sumsq)
{
  __shared__ float s1[256];
  __shared__ float s2[256];
  const int c = blockIdx.x;
  float a = 0.f, q = 0.f;
  for (int r = threadIdx.x; r < B_ * N_; r += 256) {
    float x = Wy[(size_t)r * C_ + c];
    a += x; q += x * x;
  }
  s1[threadIdx.x] = a; s2[threadIdx.x] = q;
  __syncthreads();
  for (int s = 128; s > 0; s >>= 1) {
    if (threadIdx.x < (unsigned)s) {
      s1[threadIdx.x] += s1[threadIdx.x + s];
      s2[threadIdx.x] += s2[threadIdx.x + s];
    }
    __syncthreads();
  }
  if (threadIdx.x == 0) { sums[c] = s1[0]; sumsq[c] = s2[0]; }
}

// ---------------------------------------------------------------------------
// K6: fused BatchNorm (training stats) + residual.
// ---------------------------------------------------------------------------
__global__ __launch_bounds__(256) void k_bn(
    const float* __restrict__ Wy, const float* __restrict__ vin,
    const float* __restrict__ gamma, const float* __restrict__ beta,
    const float* __restrict__ sums, const float* __restrict__ sumsq,
    float* __restrict__ out)
{
  const float inv = 1.0f / (float)(B_ * N_);
  const size_t total = (size_t)B_ * N_ * C_;
  for (size_t i = blockIdx.x * (size_t)blockDim.x + threadIdx.x; i < total;
       i += (size_t)gridDim.x * blockDim.x) {
    const int c = (int)(i & (C_ - 1));
    const float mean = sums[c] * inv;
    const float var  = sumsq[c] * inv - mean * mean;
    out[i] = (Wy[i] - mean) * rsqrtf(var + EPS_) * gamma[c] + beta[c] + vin[i];
  }
}

// ---------------------------------------------------------------------------
extern "C" void kernel_launch(void* const* d_in, const int* in_sizes, int n_in,
                              void* d_out, int out_size, void* d_ws, size_t ws_size,
                              hipStream_t stream)
{
  (void)in_sizes; (void)n_in; (void)out_size; (void)ws_size;
  const float* vin   = (const float*)d_in[0];
  const float* gw    = (const float*)d_in[1];
  const float* gb    = (const float*)d_in[2];
  const float* tw    = (const float*)d_in[3];
  const float* tb    = (const float*)d_in[4];
  const float* pw    = (const float*)d_in[5];
  const float* pb    = (const float*)d_in[6];
  const float* Ww    = (const float*)d_in[7];
  const float* Wb    = (const float*)d_in[8];
  const float* gamma = (const float*)d_in[9];
  const float* beta  = (const float*)d_in[10];

  char* ws = (char*)d_ws;
  const size_t MiB = 1024 * 1024;
  // Region A [0, 64 MiB): th | phT | gvT | ST, later overwritten by Wy (f32).
  bf16* th  = (bf16*)(ws + 0);
  bf16* phT = (bf16*)(ws + 16 * MiB);
  bf16* gvT = (bf16*)(ws + 32 * MiB);
  bf16* ST  = (bf16*)(ws + 48 * MiB);            // 2 MiB
  float* Wy = (float*)(ws + 0);                  // 64 MiB, aliases region A
  // Region B [64, 96 MiB): vbf (32 MiB, dead after k_proj); y aliases its head.
  bf16* vbf = (bf16*)(ws + 64 * MiB);
  bf16* y   = (bf16*)(ws + 64 * MiB);            // 16 MiB, written after vbf dead
  // Region C [96 MiB, ...): bf16 weights + BN stats.
  bf16* gwb = (bf16*)(ws + 96 * MiB);            // 1 MiB each
  bf16* twb = (bf16*)(ws + 97 * MiB);
  bf16* pwb = (bf16*)(ws + 98 * MiB);
  bf16* Wwb = (bf16*)(ws + 99 * MiB);
  float* sums  = (float*)(ws + 100 * MiB);
  float* sumsq = sums + C_;

  const int nV = B_ * N_ * C_ / 4;               // float4 chunks
  const int nW = IC_ * C_ / 4;
  k_cvt  <<<4096, 256, 0, stream>>>(vin, vbf, nV);
  k_cvt  <<< 512, 256, 0, stream>>>(gw, gwb, nW);
  k_cvt  <<< 512, 256, 0, stream>>>(tw, twb, nW);
  k_cvt  <<< 512, 256, 0, stream>>>(pw, pwb, nW);
  k_cvt  <<< 512, 256, 0, stream>>>(Ww, Wwb, nW);

  k_proj <<<1024, 256, 0, stream>>>(vbf, gwb, gb, twb, tb, pwb, pb, th, phT, gvT);
  k_S    <<< 256, 256, 0, stream>>>(phT, gvT, ST);
  k_y    <<<1024, 256, 0, stream>>>(th, ST, y);
  k_W    <<<2048, 256, 0, stream>>>(y, Wwb, Wb, Wy);
  k_stats<<<  C_, 256, 0, stream>>>(Wy, sums, sumsq);
  k_bn   <<<8192, 256, 0, stream>>>(Wy, vin, gamma, beta, sums, sumsq, (float*)d_out);
}
